// Attention_82205674045583
// MI455X (gfx1250) — matrix operand order repaired and checked by reading, not compiled, not run on hardware
//
#include <hip/hip_runtime.h>
#include <stdint.h>

#define B_ 64
#define T_ 4096
#define H_ 512
#define E_ 512
#define TILE_T 64
#define PADW 520  // 512 bf16 + 8 pad (16B) per row -> matches TDM pad (interval=256dw, amount=4dw)

typedef __bf16 bf16_t;
typedef bf16_t v16bf __attribute__((ext_vector_type(16)));
typedef float  v8f   __attribute__((ext_vector_type(8)));
typedef unsigned int v4u __attribute__((ext_vector_type(4)));
typedef unsigned int v2u __attribute__((ext_vector_type(2)));
typedef int    v8i   __attribute__((ext_vector_type(8)));
typedef int    v4i   __attribute__((ext_vector_type(4)));
typedef float  v4f   __attribute__((ext_vector_type(4)));
typedef float  v2f   __attribute__((ext_vector_type(2)));

#if defined(__has_builtin)
#  if __has_builtin(__builtin_amdgcn_tensor_load_to_lds)
#    define HAVE_TDM 1
#  endif
#endif
#ifndef HAVE_TDM
#  define HAVE_TDM 0
#endif

#if defined(__has_builtin) && __has_builtin(__builtin_amdgcn_s_wait_tensorcnt)
#  define WAIT_TCNT(n) __builtin_amdgcn_s_wait_tensorcnt(n)
#else
#  define WAIT_TCNT(n) asm volatile("s_wait_tensorcnt %0" ::"i"(n))
#endif

__device__ __forceinline__ unsigned short f32_to_bf16(float f) {
  union { float f; unsigned int u; } c; c.f = f;
  unsigned int u = c.u;
  unsigned int r = u + 0x7FFFu + ((u >> 16) & 1u);  // round-to-nearest-even
  return (unsigned short)(r >> 16);
}

__device__ __forceinline__ unsigned int lds_byte_off(const void* p) {
  return (unsigned int)(uintptr_t)p;  // low 32 bits of generic LDS pointer = LDS byte address
}

#if HAVE_TDM
// 2D TDM load: 16 rows x 512 bf16 slab of wt (row stride 512) into LDS with
// +16B pad after every 1024B row (so LDS row stride = PADW halfs).
__device__ __forceinline__ void tdm_load_slab(const unsigned short* wt_g,
                                              unsigned int lds_off, int nc) {
  unsigned long long ga =
      (unsigned long long)(uintptr_t)wt_g + (unsigned long long)nc * (16u * 512u * 2u);
  v4u g0;
  g0[0] = 1u;                                   // count=1, user descriptor, no gather
  g0[1] = lds_off;                              // lds_addr (bytes)
  g0[2] = (unsigned int)(ga & 0xFFFFFFFFu);     // global_addr[31:0]
  g0[3] = (unsigned int)((ga >> 32) & 0x01FFFFFFu) | 0x80000000u;  // addr[56:32] | type=2
  v8i g1;
  g1[0] = (1 << 16) | (1 << 20) | (7 << 22) | (3 << 25);  // 2B elems, pad_en, 256dw interval, 4dw pad
  g1[1] = (512 << 16);                          // tensor_dim0 = 512 (low16 at [63:48])
  g1[2] = (16 << 16);                           // tensor_dim0 hi=0; tensor_dim1 = 16
  g1[3] = (512 << 16);                          // tensor_dim1 hi=0; tile_dim0 = 512
  g1[4] = 16;                                   // tile_dim1 = 16; tile_dim2 = 0
  g1[5] = 512;                                  // tensor_dim0_stride = 512 elements
  g1[6] = 0;
  g1[7] = 0;
  v4i g2 = {0, 0, 0, 0};
  v4i g3 = {0, 0, 0, 0};
  v8i g4 = {0, 0, 0, 0, 0, 0, 0, 0};            // unused trailing group (6-arg builtin form)
  __builtin_amdgcn_tensor_load_to_lds(g0, g1, g2, g3, g4, 0);
}
#endif

// ---------------- kernel 0: W_e -> transposed bf16 (wt[n*512+k] = W_attn[(H+k)*H+n])
__global__ __launch_bounds__(256) void k_prep_wt(const float* __restrict__ W_attn,
                                                 unsigned short* __restrict__ wt) {
  int idx = blockIdx.x * 256 + threadIdx.x;  // 0..262143
  int n = idx >> 9;
  int k = idx & 511;
  wt[idx] = f32_to_bf16(W_attn[(H_ + k) * H_ + n]);
}

// ---------------- kernel 1: projb = hidden @ W_h + b_attn  (f32, tiny)
__global__ __launch_bounds__(256) void k_proj(const float* __restrict__ hidden,
                                              const float* __restrict__ W_attn,
                                              const float* __restrict__ b_attn,
                                              float* __restrict__ projb) {
  int idx = blockIdx.x * 256 + threadIdx.x;  // 0..32767
  int b = idx >> 9;
  int h = idx & 511;
  const float* hb = hidden + b * H_;
  float acc = b_attn[h];
#pragma unroll 4
  for (int k = 0; k < H_; ++k) acc += hb[k] * W_attn[k * H_ + h];
  projb[idx] = acc;
}

// ---------------- kernel 2: fused  att[b,t] = tanh(enc@W_e + projb) . v_w
__global__ __launch_bounds__(128) void k_energy(const float* __restrict__ enc,
                                                const unsigned short* __restrict__ wt,
                                                const float* __restrict__ projb,
                                                const float* __restrict__ v_w,
                                                float* __restrict__ att) {
  __shared__ unsigned short A_lds[TILE_T * PADW];     // 64 rows bf16, padded
  __shared__ unsigned short W_lds[2][16 * PADW];      // double-buffered W_e^T slabs
  __shared__ float projb_s[H_];
  __shared__ float v_s[H_];

  const int tid = threadIdx.x;
  const int b = blockIdx.x >> 6;         // 64 tiles per batch
  const int t0 = (blockIdx.x & 63) * TILE_T;

  for (int i = tid; i < H_; i += 128) {
    projb_s[i] = projb[b * H_ + i];
    v_s[i] = v_w[i];
  }

  // enc tile (64 x 512 f32, contiguous) -> bf16 LDS
  const v4f* encg = (const v4f*)(enc + ((size_t)b * T_ + t0) * (size_t)E_);
  for (int i = tid; i < TILE_T * (E_ / 4); i += 128) {
    int row = i >> 7;
    int c4 = i & 127;
    v4f x = encg[i];
    v2u p;
    p[0] = (unsigned int)f32_to_bf16(x[0]) | ((unsigned int)f32_to_bf16(x[1]) << 16);
    p[1] = (unsigned int)f32_to_bf16(x[2]) | ((unsigned int)f32_to_bf16(x[3]) << 16);
    *(v2u*)(void*)&A_lds[row * PADW + c4 * 4] = p;
  }

#if HAVE_TDM
  if (tid < 32) tdm_load_slab(wt, lds_byte_off(&W_lds[0][0]), 0);
#endif
  __syncthreads();

  const int lane = tid & 31;
  const int wave = tid >> 5;           // 0..3, 16 T-rows each
  const int half = lane >> 4;
  const int ncol = lane & 15;
  const unsigned short* aRow = &A_lds[(wave * 16 + ncol) * PADW];

  float att_acc[8];
#pragma unroll
  for (int r = 0; r < 8; ++r) att_acc[r] = 0.f;

  for (int nc = 0; nc < 32; ++nc) {
#if HAVE_TDM
    if (tid < 32) {
      if (nc + 1 < 32) {
        tdm_load_slab(wt, lds_byte_off(&W_lds[(nc + 1) & 1][0]), nc + 1);
        WAIT_TCNT(1);   // slab nc (issued earlier) complete; slab nc+1 in flight
      } else {
        WAIT_TCNT(0);
      }
    }
#else
    for (int i = tid; i < 16 * (E_ / 2); i += 128) {
      int r = i >> 8;          // /256 uints per row
      int c2 = i & 255;
      unsigned int val = ((const unsigned int*)wt)[nc * 16 * 256 + i];
      *(unsigned int*)(void*)&W_lds[nc & 1][r * PADW + c2 * 2] = val;
    }
#endif
    __syncthreads();  // slab nc ready in W_lds[nc&1]

    const unsigned short* bRow = &W_lds[nc & 1][ncol * PADW];
    v8f acc = {};
#pragma unroll
    for (int j = 0; j < 16; ++j) {
      const int k0 = j * 32;
      union { v4u u[2]; v16bf v; } fa, fb;
      // A 16x32 bf16 fragment: lanes0-15 K=[0..7]+[16..23], lanes16-31 K=[8..15]+[24..31]
      fa.u[0] = *(const v4u*)(const void*)(aRow + k0 + half * 8);
      fa.u[1] = *(const v4u*)(const void*)(aRow + k0 + 16 + half * 8);
      // B 32x16 bf16 fragment: lane=N; lanes0-15 K=[0..7]+[16..23], lanes16-31 K=[8..15]+[24..31]
      fb.u[0] = *(const v4u*)(const void*)(bRow + k0 + half * 8);
      fb.u[1] = *(const v4u*)(const void*)(bRow + k0 + 16 + half * 8);
      acc = __builtin_amdgcn_wmma_f32_16x16x32_bf16(false, fa.v, false, fb.v,
                                                    (short)0, acc, false, false);
    }

    // epilogue: energy = tanh(acc + projb[n]); att += energy * v[n]
    const float pb = projb_s[nc * 16 + ncol];
    const float vw = v_s[nc * 16 + ncol];
#pragma unroll
    for (int r = 0; r < 8; ++r) {
      float e = tanhf(acc[r] + pb);
      att_acc[r] += e * vw;
    }
    __syncthreads();  // all done reading W_lds[nc&1] before it is overwritten
  }

  // reduce over N (16 lanes per half); D row M = r + 8*half
#pragma unroll
  for (int r = 0; r < 8; ++r) {
    float red = att_acc[r];
    red += __shfl_xor(red, 8, 16);
    red += __shfl_xor(red, 4, 16);
    red += __shfl_xor(red, 2, 16);
    red += __shfl_xor(red, 1, 16);
    if (ncol == 0) att[b * T_ + t0 + wave * 16 + half * 8 + r] = red;
  }
}

// ---------------- kernel 3: softmax over T per batch, weights -> d_out
__global__ __launch_bounds__(256) void k_softmax(const float* __restrict__ att,
                                                 float* __restrict__ w_out) {
  __shared__ float red[256];
  const int b = blockIdx.x, tid = threadIdx.x;
  const float* ab = att + b * T_;
  float m = -3.4e38f;
  for (int t = tid; t < T_; t += 256) m = fmaxf(m, ab[t]);
  red[tid] = m;
  __syncthreads();
  for (int s = 128; s > 0; s >>= 1) {
    if (tid < s) red[tid] = fmaxf(red[tid], red[tid + s]);
    __syncthreads();
  }
  const float mx = red[0];
  __syncthreads();
  float sum = 0.f;
  for (int t = tid; t < T_; t += 256) sum += expf(ab[t] - mx);
  red[tid] = sum;
  __syncthreads();
  for (int s = 128; s > 0; s >>= 1) {
    if (tid < s) red[tid] += red[tid + s];
    __syncthreads();
  }
  const float inv = 1.0f / red[0];
  for (int t = tid; t < T_; t += 256) w_out[b * T_ + t] = expf(ab[t] - mx) * inv;
}

// ---------------- kernel 4: context partials over T quarters (deterministic, no atomics)
__global__ __launch_bounds__(256) void k_ctx_part(const float* __restrict__ enc,
                                                  const float* __restrict__ w,
                                                  float* __restrict__ part) {
  const int b = blockIdx.x >> 2, q = blockIdx.x & 3;
  const int e = threadIdx.x * 2;
  const float* encb = enc + ((size_t)b * T_ + (size_t)q * 1024) * E_;
  const float* wb = w + b * T_ + q * 1024;
  v2f acc = {};
  for (int t = 0; t < 1024; ++t) {
    const float ww = wb[t];
    v2f x = *(const v2f*)(encb + (size_t)t * E_ + e);
    acc[0] += ww * x[0];
    acc[1] += ww * x[1];
  }
  *(v2f*)(part + (size_t)(b * 4 + q) * E_ + e) = acc;
}

// ---------------- kernel 5: reduce partials -> context (front of d_out)
__global__ __launch_bounds__(256) void k_ctx_red(const float* __restrict__ part,
                                                 float* __restrict__ ctx) {
  int idx = blockIdx.x * 256 + threadIdx.x;  // 0..32767
  int b = idx >> 9, e = idx & 511;
  float s = 0.f;
#pragma unroll
  for (int q = 0; q < 4; ++q) s += part[(b * 4 + q) * E_ + e];
  ctx[idx] = s;
}

extern "C" void kernel_launch(void* const* d_in, const int* in_sizes, int n_in,
                              void* d_out, int out_size, void* d_ws, size_t ws_size,
                              hipStream_t stream) {
  (void)in_sizes; (void)n_in; (void)out_size; (void)ws_size;
  const float* hidden = (const float*)d_in[0];
  const float* enc    = (const float*)d_in[1];
  const float* W_attn = (const float*)d_in[2];
  const float* b_attn = (const float*)d_in[3];
  const float* v_w    = (const float*)d_in[4];

  float* ctx_out = (float*)d_out;              // (B, E)
  float* w_out   = (float*)d_out + B_ * E_;    // (B, T)

  char* ws = (char*)d_ws;
  float*          att   = (float*)ws;                                      // 1 MB
  float*          projb = (float*)(ws + (1 << 20));                        // 128 KB
  unsigned short* wt    = (unsigned short*)(ws + (1 << 20) + (128 << 10)); // 512 KB
  float*          part  = (float*)(ws + (1 << 20) + (128 << 10) + (512 << 10)); // 512 KB

  k_prep_wt<<<(512 * 512) / 256, 256, 0, stream>>>(W_attn, wt);
  k_proj<<<(B_ * H_) / 256, 256, 0, stream>>>(hidden, W_attn, b_attn, projb);
  k_energy<<<B_ * (T_ / TILE_T), 128, 0, stream>>>(enc, wt, projb, v_w, att);
  k_softmax<<<B_, 256, 0, stream>>>(att, w_out);
  k_ctx_part<<<B_ * 4, 256, 0, stream>>>(enc, w_out, part);
  k_ctx_red<<<(B_ * E_) / 256, 256, 0, stream>>>(part, ctx_out);
}
